// VolumeRenderer_15118284882227
// MI455X (gfx1250) — matrix operand order repaired and checked by reading, not compiled, not run on hardware
//
#include <hip/hip_runtime.h>

#define N_RAYS    8192
#define N_SAMPLES 256
#define HIDDEN    64
#define OUT_COLS  (3 + 1 + 1 + N_SAMPLES)   /* 261 */
#define HSTRIDE   68                        /* floats; conflict-free LDS stride (even -> b64 loads) */
#define WAVES     8

typedef __attribute__((ext_vector_type(2)))  float    v2f;
typedef __attribute__((ext_vector_type(8)))  float    v8f;
typedef __attribute__((ext_vector_type(16))) _Float16 v16h;

__device__ __forceinline__ float relu_(float x) {
  // Single v_max_num_f32 against inline-constant 0; bypasses the compiler's
  // canonicalize (v_max v,v,v) that IEEE-mode maxnum lowering inserts.
  float y;
  asm("v_max_num_f32 %0, %1, 0" : "=v"(y) : "v"(x));
  return y;
}
__device__ __forceinline__ float sigmoidf_(float x) {
  // v_rcp_f32 instead of the IEEE div_scale/div_fmas/div_fixup sequence
  return __builtin_amdgcn_rcpf(1.0f + __expf(-x));
}

__global__ __launch_bounds__(256) void volrender_wmma(
    const float* __restrict__ ray_start,   // [N,3]
    const float* __restrict__ ray_dir,     // [N,3]
    const float* __restrict__ depth,       // [N,P]
    const float* __restrict__ dists,       // [N,P]
    const int*   __restrict__ sidx,        // [N,P]
    const float* __restrict__ W1,          // [3,64]
    const float* __restrict__ b1,          // [64]
    const float* __restrict__ w_sigma,     // [64]
    const float* __restrict__ W_rgb,       // [64,3]
    const float* __restrict__ W_dir,       // [3,3]
    const float* __restrict__ b_rgb,       // [3]
    float*       __restrict__ out)         // [N, 261]
{
  __shared__ float smem_h[WAVES][16 * HSTRIDE];
  __shared__ float smem_g[WAVES][16 * 4];

  const int lane = threadIdx.x & 31;
  const int wave = threadIdx.x >> 5;
  const int ray  = blockIdx.x * WAVES + wave;
  const int m    = lane & 15;      // row (sample-in-tile) / column index
  const int hf   = lane >> 4;      // lane half selects K half per WMMA layouts

  float* hbuf = smem_h[wave];
  float* gbuf = smem_g[wave];

  // ---- Layer-1 B operand: [W1; b1] as 4x16 f32 B tiles (K=4), 4 N-tiles of HIDDEN ----
  // B layout (4x16 f32): V0 = K0 (lanes 0-15) / K2 (lanes 16-31); V1 = K1 / K3.
  v2f b1t[4];
#pragma unroll
  for (int t = 0; t < 4; ++t) {
    const int col = t * 16 + m;
    b1t[t][0] = hf ? W1[2 * HIDDEN + col] : W1[0 * HIDDEN + col];
    b1t[t][1] = hf ? b1[col]              : W1[1 * HIDDEN + col];
  }

  // ---- Layer-2 B operand: W2[64x16] (cols 0..2 = W_rgb, col 3 = w_sigma, rest 0) ----
  // f16 B 32x16, contiguous k order chosen here (matches A): VGPR j holds K = hf*16 + 2j, 2j+1 for column n = lane&15.
  v16h b2lo, b2hi;
#pragma unroll
  for (int j = 0; j < 8; ++j) {
#pragma unroll
    for (int u = 0; u < 2; ++u) {
      const int k0 = hf * 16 + 2 * j + u;       // tile 0: K 0..31
      const int k1 = k0 + 32;                   // tile 1: K 32..63
      float w0 = 0.f, w1 = 0.f;
      if (m < 3)       { w0 = W_rgb[k0 * 3 + m]; w1 = W_rgb[k1 * 3 + m]; }
      else if (m == 3) { w0 = w_sigma[k0];       w1 = w_sigma[k1]; }
      b2lo[2 * j + u] = (_Float16)w0;
      b2hi[2 * j + u] = (_Float16)w1;
    }
  }

  // ---- Per-ray constants ----
  const float rsx = ray_start[ray * 3 + 0];
  const float rsy = ray_start[ray * 3 + 1];
  const float rsz = ray_start[ray * 3 + 2];
  const float rdx = ray_dir[ray * 3 + 0];
  const float rdy = ray_dir[ray * 3 + 1];
  const float rdz = ray_dir[ray * 3 + 2];
  float dirW[3];
#pragma unroll
  for (int c = 0; c < 3; ++c)
    dirW[c] = rdx * W_dir[0 * 3 + c] + rdy * W_dir[1 * 3 + c] + rdz * W_dir[2 * 3 + c] + b_rgb[c];

  float T = 0.f;                               // running cumsum(free_energy)
  float accR = 0.f, accG = 0.f, accB = 0.f, accD = 0.f, accP = 0.f;

  // ---- software-pipelined per-sample loads (one tile ahead, branchless) ----
  const int base = ray * N_SAMPLES;
  float d_cur    = depth[base + m];
  float dist_cur = dists[base + m];
  int   vid_cur  = sidx[base + m];

  for (int it = 0; it < 16; ++it) {
    const int s  = it * 16 + m;                // sample index (lanes 16-31 mirror 0-15)
    const int sn = s + ((it < 15) ? 16 : 0);   // clamped next-tile index (branchless)
    const float d_nxt    = depth[base + sn];   // issued early; consumed next iteration
    const float dist_nxt = dists[base + sn];
    const int   vid_nxt  = sidx[base + sn];

    const float d = d_cur;

    // ---- Stage 1: h = relu([x y z 1] @ [W1;b1]) via 4x V_WMMA_F32_16X16X4_F32 ----
    // A layout (16x4 f32): V0 = K0/K2, V1 = K1/K3 by lane half.
    v2f a1;
    a1[0] = hf ? (rsz + rdz * d) : (rsx + rdx * d);
    a1[1] = hf ? 1.0f            : (rsy + rdy * d);

    v8f h[4];
#pragma unroll
    for (int t = 0; t < 4; ++t) {
      v8f c = {};
      c = __builtin_amdgcn_wmma_f32_16x16x4_f32(false, a1, false, b1t[t],
                                                (short)0, c, false, false);
      h[t] = c;
    }

    // relu + scatter h (C layout) to row-major LDS h[row][col]
#pragma unroll
    for (int t = 0; t < 4; ++t) {
#pragma unroll
      for (int r = 0; r < 8; ++r) {
        const float v = relu_(h[t][r]);
        const int row = hf ? (r + 8) : r;
        hbuf[row * HSTRIDE + t * 16 + m] = v;
      }
    }
    asm volatile("s_wait_dscnt 0" ::: "memory");

    // ---- Stage 2: g = h @ W2 via 2x V_WMMA_F32_16X16X32_F16 (K=64) ----
    // f16 A 16x32, contiguous k order chosen here (matches B): V j(0..7): K = hf*16 + 2j.
    v16h a2lo, a2hi;
    const v2f* hr = (const v2f*)&hbuf[m * HSTRIDE];   // pair index = k/2 (8B-aligned)
#pragma unroll
    for (int j = 0; j < 8; ++j) {
      const int k = hf * 16 + 2 * j;
      const v2f plo = hr[k >> 1];
      const v2f phi = hr[(k + 32) >> 1];
      a2lo[2 * j]     = (_Float16)plo[0];
      a2lo[2 * j + 1] = (_Float16)plo[1];
      a2hi[2 * j]     = (_Float16)phi[0];
      a2hi[2 * j + 1] = (_Float16)phi[1];
    }
    v8f g = {};
    g = __builtin_amdgcn_wmma_f32_16x16x32_f16(false, a2lo, false, b2lo, (short)0, g, false, false);
    g = __builtin_amdgcn_wmma_f32_16x16x32_f16(false, a2hi, false, b2hi, (short)0, g, false, false);

    // stage D cols 0..3 (rgb logits, sigma) so lane s owns sample s
    if (m < 4) {
#pragma unroll
      for (int r = 0; r < 8; ++r) {
        const int row = hf ? (r + 8) : r;
        gbuf[row * 4 + m] = g[r];
      }
    }
    asm volatile("s_wait_dscnt 0" ::: "memory");

    const float l0 = gbuf[m * 4 + 0];
    const float l1 = gbuf[m * 4 + 1];
    const float l2 = gbuf[m * 4 + 2];
    const float sg = gbuf[m * 4 + 3];

    // ---- free energy + masked scatter ----
    const float sigp = relu_(sg);
    const float fe   = (vid_cur != -1) ? (sigp * dist_cur * 7.0f) : 0.f;

    // inclusive prefix over the 16-sample group (mirror-safe across lane halves)
    float pre = fe;
#pragma unroll
    for (int off = 1; off < 16; off <<= 1) {
      const float v = __shfl_up(pre, off);
      if (m >= off) pre += v;
    }
    const float excl = pre - fe;
    const float p = (1.f - __expf(-fe)) * __expf(-(T + excl));

    const float cr = sigmoidf_(l0 + dirW[0]);
    const float cg = sigmoidf_(l1 + dirW[1]);
    const float cb = sigmoidf_(l2 + dirW[2]);

    accR += cr * p;
    accG += cg * p;
    accB += cb * p;
    accD += d * p;
    accP += p;

    if (hf == 0) out[ray * OUT_COLS + 5 + s] = p;

    T += __shfl(pre, 15);                      // carry tile total (lane 15 inclusive)

    d_cur = d_nxt; dist_cur = dist_nxt; vid_cur = vid_nxt;
  }

  // ---- reduce per-lane partials across the 16-sample group ----
#pragma unroll
  for (int off = 8; off >= 1; off >>= 1) {
    accR += __shfl_xor(accR, off);
    accG += __shfl_xor(accG, off);
    accB += __shfl_xor(accB, off);
    accD += __shfl_xor(accD, off);
    accP += __shfl_xor(accP, off);
  }
  if (lane == 0) {
    out[ray * OUT_COLS + 0] = accR;
    out[ray * OUT_COLS + 1] = accG;
    out[ray * OUT_COLS + 2] = accB;
    out[ray * OUT_COLS + 3] = accD;
    out[ray * OUT_COLS + 4] = 1.f - accP;
  }
}

extern "C" void kernel_launch(void* const* d_in, const int* in_sizes, int n_in,
                              void* d_out, int out_size, void* d_ws, size_t ws_size,
                              hipStream_t stream) {
  const float* ray_start = (const float*)d_in[0];
  const float* ray_dir   = (const float*)d_in[1];
  const float* depth     = (const float*)d_in[2];
  const float* dists     = (const float*)d_in[3];
  const int*   sidx      = (const int*)  d_in[4];
  const float* W1        = (const float*)d_in[5];
  const float* b1        = (const float*)d_in[6];
  const float* w_sigma   = (const float*)d_in[7];
  const float* W_rgb     = (const float*)d_in[8];
  const float* W_dir     = (const float*)d_in[9];
  const float* b_rgb     = (const float*)d_in[10];
  float* out = (float*)d_out;

  dim3 grid(N_RAYS / WAVES), block(32 * WAVES);
  volrender_wmma<<<grid, block, 0, stream>>>(ray_start, ray_dir, depth, dists, sidx,
                                             W1, b1, w_sigma, W_rgb, W_dir, b_rgb, out);
}
